// Attention_65515431133747
// MI455X (gfx1250) — matrix operand order repaired and checked by reading, not compiled, not run on hardware
//
#include <hip/hip_runtime.h>

typedef __attribute__((ext_vector_type(16))) __bf16 v16bf;
typedef __attribute__((ext_vector_type(8)))  float  v8f;
typedef unsigned int uint32;

union FragB16 {
  uint4 q[2];
  v16bf v;
};

__device__ __forceinline__ uint32 pack_bf16(float a, float b) {
  union { float f; uint32 u; } ua, ub;
  ua.f = a; ub.f = b;
  uint32 x = ua.u + 0x7FFFu + ((ua.u >> 16) & 1u);  // round-to-nearest-even
  uint32 y = ub.u + 0x7FFFu + ((ub.u >> 16) & 1u);
  return (x >> 16) | (y & 0xFFFF0000u);
}

// tanh via hardware exp+rcp: ~6 VALU ops, exact at saturation.
__device__ __forceinline__ float fast_tanh(float x) {
  float e = __expf(2.0f * x);          // v_exp_f32
  return 1.0f - 2.0f / (e + 1.0f);     // v_rcp_f32 path
}

// ---------------------------------------------------------------------------
// One-time: weight [512,512] fp32 (row-major [k][n]) -> bf16 tiles in WMMA
// 16x16x32 B-fragment order. Tile (kb, nt): 32 lanes x 16 bf16; lane holds
// column n = nt*16 + lane%16, K = kb*32 + (lane<16?0:16) + j, j ascending.
// Per-lane data = 32 contiguous bytes -> two global_load_b128 in the GEMM.
// ---------------------------------------------------------------------------
__global__ __launch_bounds__(256) void k_prep_weight(
    const float* __restrict__ w, uint4* __restrict__ wbf)
{
  constexpr int HID = 512;
  const int gid  = blockIdx.x * 256 + threadIdx.x;   // 0..16383
  const int lane = gid & 31;
  const int nt   = (gid >> 5) & 31;
  const int kb   = gid >> 10;
  const int n     = nt * 16 + (lane & 15);
  const int kbase = kb * 32 + ((lane & 16) ? 16 : 0);
  uint32 pk[8];
#pragma unroll
  for (int j = 0; j < 8; ++j) {
    float a = w[(size_t)(kbase + 2 * j)     * HID + n];
    float b = w[(size_t)(kbase + 2 * j + 1) * HID + n];
    pk[j] = pack_bf16(a, b);
  }
  uint4* dst = wbf + (size_t)gid * 2;
  dst[0] = make_uint4(pk[0], pk[1], pk[2], pk[3]);
  dst[1] = make_uint4(pk[4], pk[5], pk[6], pk[7]);
}

// ---------------------------------------------------------------------------
// Fused: scores[b,t] = tanh( tanh(emb[b,t,:] @ W + bias) @ cw )
// Workgroup = 64 t-rows x 512 N cols. 8 waves; each wave: 4 M-tiles x 4
// N-tiles = 16 WMMA accumulators. K loop: 16 steps of 32, LDS A staging
// double-buffered (one barrier per step). u never touches memory.
// B re-read per WG drops 4x vs M=16 (1 GB total L2 traffic).
// ---------------------------------------------------------------------------
__global__ __launch_bounds__(256) void k_scores(
    const float* __restrict__ emb,   // [B,T,HID] fp32
    const uint4* __restrict__ wbf,   // bf16 fragment-ordered weight
    const float* __restrict__ bias,  // [HID]
    const float* __restrict__ cw,    // [HID]
    float* __restrict__ scores)      // [B,T]
{
  constexpr int T = 2048, HID = 512;
  const int tid  = threadIdx.x;
  const int lane = tid & 31;
  const int wave = tid >> 5;
  const int b  = blockIdx.x >> 5;          // T/64 = 32 tiles per batch row
  const int t0 = (blockIdx.x & 31) << 6;

  __shared__ uint32 sAu[2][1024];          // 2 x (64x32 bf16) A tiles
  __shared__ float  sScore[64];
  if (tid < 64) sScore[tid] = 0.0f;

  const float* embT = emb + ((size_t)b * T + t0) * HID;

  // Stage one 64x32 fp32 chunk as bf16 pairs in A-fragment order.
  // pair index p: mt = p>>8, lane = (p>>3)&31, j0 = (p&7)*2;
  // M = mt*16 + lane%16; K = (lane<16?0:16) + j0.
  auto stage = [&](int kb, int buf) {
#pragma unroll
    for (int i = 0; i < 4; ++i) {
      const int p  = tid + i * 256;
      const int mt = p >> 8;
      const int ln = (p >> 3) & 31;
      const int j0 = (p & 7) << 1;
      const int Mg = mt * 16 + (ln & 15);
      const int K0 = ((ln & 16) ? 16 : 0) + j0;
      float2 av = *(const float2*)(embT + (size_t)Mg * HID + kb * 32 + K0);
      sAu[buf][p] = pack_bf16(av.x, av.y);
    }
  };

  v8f acc[4][4];
#pragma unroll
  for (int m = 0; m < 4; ++m)
#pragma unroll
    for (int c = 0; c < 4; ++c) acc[m][c] = v8f{};

  stage(0, 0);

  for (int kb = 0; kb < 16; ++kb) {
    __syncthreads();                       // buffer kb ready; kb-1 consumed
    if (kb + 1 < 16) stage(kb + 1, (kb + 1) & 1);

    FragB16 bf[4];
#pragma unroll
    for (int c = 0; c < 4; ++c) {          // B frags, L2-resident
      const uint4* bp =
          wbf + ((size_t)((kb * 32 + wave * 4 + c) * 32 + lane)) * 2;
      bf[c].q[0] = bp[0];
      bf[c].q[1] = bp[1];
    }

    const uint32* abase = &sAu[kb & 1][0];
#pragma unroll
    for (int m = 0; m < 4; ++m) {
      FragB16 af;
      const uint4* ap = (const uint4*)(abase + (m * 32 + lane) * 8);
      af.q[0] = ap[0];                     // ds_load_b128 x2
      af.q[1] = ap[1];
#pragma unroll
      for (int c = 0; c < 4; ++c)
        acc[m][c] = __builtin_amdgcn_wmma_f32_16x16x32_bf16(
            false, af.v, false, bf[c].v, (short)0, acc[m][c], false, false);
    }
  }

  // Epilogue: u = tanh(acc + bias[n]); partial score = sum_n u*cw[n].
  // C layout: VGPR r, lane l -> M = mt*16 + r + (l>=16?8:0), N = n0 + l%16.
  float part[4][8];
#pragma unroll
  for (int m = 0; m < 4; ++m)
#pragma unroll
    for (int r = 0; r < 8; ++r) part[m][r] = 0.0f;

#pragma unroll
  for (int c = 0; c < 4; ++c) {
    const int n = wave * 64 + c * 16 + (lane & 15);
    const float bi  = bias[n];
    const float cwn = cw[n];
#pragma unroll
    for (int m = 0; m < 4; ++m)
#pragma unroll
      for (int r = 0; r < 8; ++r)
        part[m][r] += fast_tanh(acc[m][c][r] + bi) * cwn;
  }

#pragma unroll
  for (int m = 0; m < 4; ++m)
#pragma unroll
    for (int r = 0; r < 8; ++r) {          // reduce over 16-lane N groups
      float v = part[m][r];
      v += __shfl_xor(v, 1);
      v += __shfl_xor(v, 2);
      v += __shfl_xor(v, 4);
      v += __shfl_xor(v, 8);
      if ((lane & 15) == 0)
        atomicAdd(&sScore[m * 16 + r + ((lane & 16) ? 8 : 0)], v); // ds_add_f32
    }
  __syncthreads();
  if (tid < 64)
    scores[(size_t)b * T + t0 + tid] = fast_tanh(sScore[tid]);
}

// ---------------------------------------------------------------------------
// Softmax over T per batch row (one 256-thread block per row).
// ---------------------------------------------------------------------------
__global__ __launch_bounds__(256) void k_softmax(
    const float* __restrict__ scores, float* __restrict__ attn)
{
  constexpr int T = 2048;
  const int b = blockIdx.x, tid = threadIdx.x;
  __shared__ float red[256];
  const float* s = scores + (size_t)b * T;

  float v[8], mx = -3.402823466e38f;
#pragma unroll
  for (int i = 0; i < 8; ++i) { v[i] = s[tid + i * 256]; mx = fmaxf(mx, v[i]); }
  red[tid] = mx; __syncthreads();
  for (int st = 128; st > 0; st >>= 1) {
    if (tid < st) red[tid] = fmaxf(red[tid], red[tid + st]);
    __syncthreads();
  }
  const float m = red[0];
  __syncthreads();
  float e[8], sum = 0.0f;
#pragma unroll
  for (int i = 0; i < 8; ++i) { e[i] = __expf(v[i] - m); sum += e[i]; }
  red[tid] = sum; __syncthreads();
  for (int st = 128; st > 0; st >>= 1) {
    if (tid < st) red[tid] += red[tid + st];
    __syncthreads();
  }
  const float inv = 1.0f / red[0];
#pragma unroll
  for (int i = 0; i < 8; ++i) attn[(size_t)b * T + tid + i * 256] = e[i] * inv;
}

// ---------------------------------------------------------------------------
// ctx[b,h] = sum_t emb[b,t,h] * attn[b,t] — streaming, HBM-bound.
// ---------------------------------------------------------------------------
__global__ __launch_bounds__(256) void k_ctx(
    const float* __restrict__ emb, const float* __restrict__ attn,
    float* __restrict__ ctx)
{
  constexpr int T = 2048, HID = 512;
  const int b = blockIdx.x >> 1;
  const int h = ((blockIdx.x & 1) << 8) + threadIdx.x;
  const float* e = emb + (size_t)b * T * HID + h;
  const float* a = attn + (size_t)b * T;
  float acc = 0.0f;
#pragma unroll 4
  for (int t = 0; t < T; ++t) acc += e[(size_t)t * HID] * a[t];
  ctx[b * HID + h] = acc;
}

// ---------------------------------------------------------------------------
// h = relu(ctx @ W1 + b1). W1 access coalesced across c.
// ---------------------------------------------------------------------------
__global__ __launch_bounds__(256) void k_fc1(
    const float* __restrict__ ctx, const float* __restrict__ W1,
    const float* __restrict__ b1, float* __restrict__ hout)
{
  constexpr int HID = 512, HC = 1024;
  const int idx = blockIdx.x * 256 + threadIdx.x;   // 65536 total
  const int r = idx >> 10, c = idx & (HC - 1);
  const float* x = ctx + r * HID;
  float acc = b1[c];
#pragma unroll 4
  for (int k = 0; k < HID; ++k) acc += x[k] * W1[(size_t)k * HC + c];
  hout[idx] = fmaxf(acc, 0.0f);
}

// ---------------------------------------------------------------------------
// logits = h @ W2 + b2 ; out = softmax(logits). One block per row, tiny.
// ---------------------------------------------------------------------------
__global__ __launch_bounds__(256) void k_fc2(
    const float* __restrict__ h, const float* __restrict__ W2,
    const float* __restrict__ b2, float* __restrict__ out)
{
  constexpr int HC = 1024, DO = 10;
  const int r = blockIdx.x, tid = threadIdx.x;
  __shared__ float red[256];
  __shared__ float logit[DO];
  const float* x = h + (size_t)r * HC;
  for (int j = 0; j < DO; ++j) {
    float acc = 0.0f;
    for (int k = tid; k < HC; k += 256) acc += x[k] * W2[(size_t)k * DO + j];
    red[tid] = acc; __syncthreads();
    for (int st = 128; st > 0; st >>= 1) {
      if (tid < st) red[tid] += red[tid + st];
      __syncthreads();
    }
    if (tid == 0) logit[j] = red[0] + b2[j];
    __syncthreads();
  }
  if (tid == 0) {
    float m = logit[0];
    for (int j = 1; j < DO; ++j) m = fmaxf(m, logit[j]);
    float e[DO], S = 0.0f;
    for (int j = 0; j < DO; ++j) { e[j] = __expf(logit[j] - m); S += e[j]; }
    const float inv = 1.0f / S;
    for (int j = 0; j < DO; ++j) out[(size_t)r * DO + j] = e[j] * inv;
  }
}

extern "C" void kernel_launch(void* const* d_in, const int* in_sizes, int n_in,
                              void* d_out, int out_size, void* d_ws, size_t ws_size,
                              hipStream_t stream) {
  (void)in_sizes; (void)n_in; (void)out_size; (void)ws_size;
  // inputs: 0=numerical(unused) 1=embedding 2=weight 3=bias 4=context_weight
  //         5=W1 6=b1 7=W2 8=b2
  const float* emb  = (const float*)d_in[1];
  const float* w    = (const float*)d_in[2];
  const float* bias = (const float*)d_in[3];
  const float* cw   = (const float*)d_in[4];
  const float* W1   = (const float*)d_in[5];
  const float* b1   = (const float*)d_in[6];
  const float* W2   = (const float*)d_in[7];
  const float* b2   = (const float*)d_in[8];
  float* out = (float*)d_out;

  char* ws = (char*)d_ws;
  uint4* wbf    = (uint4*)(ws);                  // 512 KB bf16 weight frags
  float* scores = (float*)(ws + (512 << 10));    // 512 KB [B,T]
  float* attn   = (float*)(ws + (1024 << 10));   // 512 KB [B,T]
  float* ctx    = (float*)(ws + (1536 << 10));   // 128 KB [B,HID]
  float* hbuf   = (float*)(ws + (1664 << 10));   // 256 KB [B,H_CLS]

  k_prep_weight<<<64,      256, 0, stream>>>(w, wbf);
  k_scores     <<<64 * 32, 256, 0, stream>>>(emb, wbf, bias, cw, scores);
  k_softmax    <<<64,      256, 0, stream>>>(scores, attn);
  k_ctx        <<<128,     256, 0, stream>>>(emb, attn, ctx);
  k_fc1        <<<256,     256, 0, stream>>>(ctx, W1, b1, hbuf);
  k_fc2        <<<64,      256, 0, stream>>>(hbuf, W2, b2, out);
}
